// ArielEncoderCell2_23587960390073
// MI455X (gfx1250) — matrix operand order repaired and checked by reading, not compiled, not run on hardware
//
#include <hip/hip_runtime.h>
#include <hip/hip_bf16.h>
#include <stddef.h>

// ---------------------------------------------------------------------------
// ArielEncoderCell2: 32-step LSTM arithmetic-coding encoder for MI455X/gfx1250
//   VOCAB=2048, EMB=256, LAT=64, MAXLEN=32, BATCH=64
//   bf16 WMMA (v_wmma_f32_16x16x32_bf16), fp32 accumulate.
//   Weights converted+packed once per call to a B-fragment-contiguous bf16
//   layout (L2-resident, 36 MB); activations staged fp32->bf16 through a
//   double-buffered LDS tile (one barrier per K-chunk).
// ---------------------------------------------------------------------------

typedef __attribute__((ext_vector_type(8)))  float   v8f;
typedef __attribute__((ext_vector_type(16))) __bf16  bf16x16;
typedef __attribute__((ext_vector_type(8)))  __bf16  bf16x8;

#define BATCH  64
#define VOCAB  2048
#define EMB    256
#define LAT    64
#define MAXLEN 32
#define NGATE  8192   // 4 * VOCAB
#define CHSTRIDE ((size_t)NGATE * 32)   // packed elements per K-chunk

__device__ __forceinline__ float sigm(float x) { return 1.0f / (1.0f + expf(-x)); }

// ---------------------------------------------------------------------------
// Init: zero h0, c; low=0, upp=3.0
// ---------------------------------------------------------------------------
__global__ __launch_bounds__(256) void aec_init(float* __restrict__ h0,
                                                float* __restrict__ c,
                                                float* __restrict__ low,
                                                float* __restrict__ upp) {
    int i = blockIdx.x * 256 + threadIdx.x;
    if (i < BATCH * VOCAB) { h0[i] = 0.0f; c[i] = 0.0f; }
    if (i < BATCH * LAT)   { low[i] = 0.0f; upp[i] = 3.0f; }
}

// ---------------------------------------------------------------------------
// Pack: W[K x 8192] fp32 row-major -> Wp bf16 layout [kc][col][ki] (kc=K/32).
// Per-lane WMMA B fragment (16 consecutive K of one column) becomes one
// contiguous 32B read. grid.x = (K/32) * 128 ; tile = 32k x 64n.
// ---------------------------------------------------------------------------
__global__ __launch_bounds__(256) void aec_pack(const float* __restrict__ W,
                                                __bf16* __restrict__ Wp) {
    const int tid = threadIdx.x;
    const int tn  = blockIdx.x & 127;    // column tile (64 cols)
    const int tk  = blockIdx.x >> 7;     // k chunk (32 rows)
    const int k0 = tk * 32, n0 = tn * 64;
    __shared__ __bf16 tile[32][65];
    {
        int rr = tid >> 3, cb = (tid & 7) * 8;
        const float* src = W + (size_t)(k0 + rr) * NGATE + n0 + cb;
#pragma unroll
        for (int j = 0; j < 8; ++j) tile[rr][cb + j] = (__bf16)src[j];
    }
    __syncthreads();
    {
        int nl = tid >> 2, ki = (tid & 3) * 8;
        __bf16* dst = Wp + ((size_t)tk * NGATE + (n0 + nl)) * 32 + ki;
#pragma unroll
        for (int j = 0; j < 8; ++j) dst[j] = tile[ki + j][nl];   // fully coalesced
    }
}

// ---------------------------------------------------------------------------
// Step A: softmax stats of h, arithmetic-coding bound update, emit z[:, t, :].
// One block per batch row.
// ---------------------------------------------------------------------------
__global__ __launch_bounds__(256) void aec_stepA(const int*   __restrict__ tokens,
                                                 const float* __restrict__ h,
                                                 float* __restrict__ low,
                                                 float* __restrict__ upp,
                                                 float* __restrict__ z,
                                                 int t) {
    const int b = blockIdx.x, tid = threadIdx.x;
    const int tok = tokens[b * MAXLEN + t];
    const float* hb = h + (size_t)b * VOCAB;

    __shared__ float red[256];
    __shared__ float sh_m, sh_S, sh_Sle, sh_p, sh_clow, sh_cupp;

    float m = -3.4e38f;
    for (int v = tid; v < VOCAB; v += 256) m = fmaxf(m, hb[v]);
    red[tid] = m; __syncthreads();
    for (int s = 128; s > 0; s >>= 1) {
        if (tid < s) red[tid] = fmaxf(red[tid], red[tid + s]);
        __syncthreads();
    }
    if (tid == 0) sh_m = red[0];
    __syncthreads();
    m = sh_m;

    float S = 0.0f, Sle = 0.0f;
    for (int v = tid; v < VOCAB; v += 256) {
        float e = expf(hb[v] - m);
        S += e;
        if (v <= tok) Sle += e;
        if (v == tok) sh_p = e;    // exactly one writer in the block
    }
    red[tid] = S; __syncthreads();
    for (int s = 128; s > 0; s >>= 1) { if (tid < s) red[tid] += red[tid + s]; __syncthreads(); }
    if (tid == 0) sh_S = red[0];
    __syncthreads();
    red[tid] = Sle; __syncthreads();
    for (int s = 128; s > 0; s >>= 1) { if (tid < s) red[tid] += red[tid + s]; __syncthreads(); }
    if (tid == 0) sh_Sle = red[0];
    __syncthreads();

    if (tid == 0) {
        float invS = 1.0f / sh_S;
        float cu = sh_Sle * invS;
        sh_cupp = cu;
        sh_clow = cu - sh_p * invS;
    }
    __syncthreads();

    if (tid < LAT) {
        float lo = low[b * LAT + tid];
        float up = upp[b * LAT + tid];
        if (tid == (t & (LAT - 1))) {
            float rng = up - lo;
            float nlo = lo + rng * sh_clow;
            float nup = lo + rng * sh_cupp;
            low[b * LAT + tid] = nlo;
            upp[b * LAT + tid] = nup;
            lo = nlo; up = nup;
        }
        z[((size_t)b * MAXLEN + t) * LAT + tid] = 0.5f * (lo + up);
    }
}

// ---------------------------------------------------------------------------
// Step B: gates = h@Wh + emb[tok]@Wx + b ; LSTM pointwise ; write h_out, c.
// 128 blocks x 256 threads (8 waves). Block owns 16 hidden units (all 4 gates,
// all 64 batch rows). wave: g = wave&3 (i,f,g,o); waves 0-3 rows 0-31,
// waves 4-7 rows 32-63 (two 16x16 accumulators per wave sharing one B frag).
// Split K phases (Wh: 64 chunks, Wx: 8 chunks) with double-buffered LDS A.
// ---------------------------------------------------------------------------
__global__ __launch_bounds__(256) void aec_stepB(const float* __restrict__ h_in,
                                                 float* __restrict__ c_st,
                                                 const __bf16* __restrict__ Whp,
                                                 const __bf16* __restrict__ Wxp,
                                                 const float* __restrict__ bias,
                                                 const float* __restrict__ emb,
                                                 const int*   __restrict__ tokens,
                                                 float* __restrict__ h_out,
                                                 int t) {
    const int tid  = threadIdx.x;
    const int lane = tid & 31;
    const int wave = tid >> 5;
    const int g    = wave & 3;            // gate: 0=i 1=f 2=g 3=o
    const int rt   = (wave >> 2) * 2;     // row tiles rt, rt+1 (16 rows each)
    const int nc   = blockIdx.x * 16;     // hidden-unit tile

    __shared__ __bf16 lds_a[2][64][40];   // double-buffered A chunk; stride 40
                                          // -> conflict-free + 16B-aligned rows
    __shared__ float  lds_z[4][64][17];   // gate pre-activations
    __shared__ int    lds_tok[64];

    if (tid < 64) lds_tok[tid] = tokens[tid * MAXLEN + t];

    const int col = g * VOCAB + nc + (lane & 15);     // gate column in [0,8192)
    const int kbB = (lane & 16) ? 16 : 0;             // B frag: 16 consecutive K
    const int kbA = (lane & 16) ? 16 : 0;             // A frag: 16 consecutive K
    const int m0  = rt * 16 + (lane & 15);

    v8f acc0 = {};
    v8f acc1 = {};

    const int r  = tid >> 2;              // staging: row 0..63
    const int kk = (tid & 3) * 8;         // staging: 8 consecutive k

    // Prime buffer 0 with Wh K-chunk 0
    {
        const float* s0 = h_in + (size_t)r * VOCAB + kk;
#pragma unroll
        for (int j = 0; j < 8; ++j) lds_a[0][r][kk + j] = (__bf16)s0[j];
    }
    __syncthreads();

    const __bf16* bph = Whp + (size_t)col * 32 + kbB;
    const __bf16* bpx = Wxp + (size_t)col * 32 + kbB;

    // ---- Phase 1: Wh (64 chunks of K=32) --------------------------------
    for (int chunk = 0; chunk < 64; ++chunk) {
        const int p = chunk & 1;
        bf16x16 bfrag = *(const bf16x16*)(bph + (size_t)chunk * CHSTRIDE);

        bf16x8 alo0 = *(const bf16x8*)&lds_a[p][m0][kbA];
        bf16x8 ahi0 = *(const bf16x8*)&lds_a[p][m0][kbA + 8];
        bf16x8 alo1 = *(const bf16x8*)&lds_a[p][m0 + 16][kbA];
        bf16x8 ahi1 = *(const bf16x8*)&lds_a[p][m0 + 16][kbA + 8];

        // Stage next chunk into the other buffer (chunk 63 pre-stages Wx chunk 0)
        const float* s = (chunk < 63)
            ? (h_in + (size_t)r * VOCAB + (chunk + 1) * 32 + kk)
            : (emb + (size_t)lds_tok[r] * EMB + kk);
#pragma unroll
        for (int j = 0; j < 8; ++j) lds_a[p ^ 1][r][kk + j] = (__bf16)s[j];

        bf16x16 a0, a1;
#pragma unroll
        for (int j = 0; j < 8; ++j) {
            a0[j] = alo0[j]; a0[j + 8] = ahi0[j];
            a1[j] = alo1[j]; a1[j + 8] = ahi1[j];
        }
        __syncthreads();
        acc0 = __builtin_amdgcn_wmma_f32_16x16x32_bf16(false, a0, false, bfrag,
                                                       (short)0, acc0, false, false);
        acc1 = __builtin_amdgcn_wmma_f32_16x16x32_bf16(false, a1, false, bfrag,
                                                       (short)0, acc1, false, false);
    }

    // ---- Phase 2: Wx (8 chunks of K=32), x = emb[token] -----------------
    for (int chunk = 0; chunk < 8; ++chunk) {
        const int p = chunk & 1;          // Wx chunk 0 landed in buffer 0
        bf16x16 bfrag = *(const bf16x16*)(bpx + (size_t)chunk * CHSTRIDE);

        bf16x8 alo0 = *(const bf16x8*)&lds_a[p][m0][kbA];
        bf16x8 ahi0 = *(const bf16x8*)&lds_a[p][m0][kbA + 8];
        bf16x8 alo1 = *(const bf16x8*)&lds_a[p][m0 + 16][kbA];
        bf16x8 ahi1 = *(const bf16x8*)&lds_a[p][m0 + 16][kbA + 8];

        if (chunk < 7) {
            const float* s = emb + (size_t)lds_tok[r] * EMB + (chunk + 1) * 32 + kk;
#pragma unroll
            for (int j = 0; j < 8; ++j) lds_a[p ^ 1][r][kk + j] = (__bf16)s[j];
        }

        bf16x16 a0, a1;
#pragma unroll
        for (int j = 0; j < 8; ++j) {
            a0[j] = alo0[j]; a0[j + 8] = ahi0[j];
            a1[j] = alo1[j]; a1[j + 8] = ahi1[j];
        }
        __syncthreads();
        acc0 = __builtin_amdgcn_wmma_f32_16x16x32_bf16(false, a0, false, bfrag,
                                                       (short)0, acc0, false, false);
        acc1 = __builtin_amdgcn_wmma_f32_16x16x32_bf16(false, a1, false, bfrag,
                                                       (short)0, acc1, false, false);
    }

    // C/D layout: lane L -> N = L%16 ; VGPR v -> M = v + (L>=16 ? 8 : 0)
    const float bv = bias[col];
    const int mb = rt * 16 + ((lane & 16) ? 8 : 0);
    const int n  = lane & 15;
#pragma unroll
    for (int v = 0; v < 8; ++v) {
        lds_z[g][mb + v][n]      = acc0[v] + bv;
        lds_z[g][mb + 16 + v][n] = acc1[v] + bv;
    }
    __syncthreads();

    // Fused LSTM pointwise for this block's 16 hidden units x 64 rows
    for (int idx = tid; idx < 64 * 16; idx += 256) {
        int mrow = idx >> 4, nn = idx & 15;
        int hu = nc + nn;
        float zi = lds_z[0][mrow][nn];
        float zf = lds_z[1][mrow][nn];
        float zg = lds_z[2][mrow][nn];
        float zo = lds_z[3][mrow][nn];
        float cold = c_st[(size_t)mrow * VOCAB + hu];
        float cn = sigm(zf) * cold + sigm(zi) * tanhf(zg);
        c_st[(size_t)mrow * VOCAB + hu]  = cn;
        h_out[(size_t)mrow * VOCAB + hu] = sigm(zo) * tanhf(cn);
    }
}

// ---------------------------------------------------------------------------
// kernel_launch
//   inputs: tokens(i32 64x32), emb(f32 2048x256), Wx(f32 256x8192),
//           Wh(f32 2048x8192), b(f32 8192)
//   output: z f32 [64][32][64]
// ---------------------------------------------------------------------------
extern "C" void kernel_launch(void* const* d_in, const int* in_sizes, int n_in,
                              void* d_out, int out_size, void* d_ws, size_t ws_size,
                              hipStream_t stream) {
    const int*   tokens = (const int*)d_in[0];
    const float* emb    = (const float*)d_in[1];
    const float* Wx     = (const float*)d_in[2];
    const float* Wh     = (const float*)d_in[3];
    const float* bias   = (const float*)d_in[4];
    float* out = (float*)d_out;

    char* ws = (char*)d_ws;
    // workspace layout (~37.6 MB total)
    __bf16* Whp = (__bf16*)(ws + 0);                       // 32 MB
    __bf16* Wxp = (__bf16*)(ws + (size_t)33554432);        //  4 MB
    float*  h0  = (float*)(ws + (size_t)37748736);         // 512 KB
    float*  h1  = (float*)(ws + (size_t)38273024);         // 512 KB
    float*  c   = (float*)(ws + (size_t)38797312);         // 512 KB
    float*  low = (float*)(ws + (size_t)39321600);         //  16 KB
    float*  upp = (float*)(ws + (size_t)39337984);         //  16 KB

    aec_init<<<512, 256, 0, stream>>>(h0, c, low, upp);
    aec_pack<<<64 * 128, 256, 0, stream>>>(Wh, Whp);       // 2048 K rows
    aec_pack<<<8 * 128, 256, 0, stream>>>(Wx, Wxp);        //  256 K rows

    float* hbuf[2] = { h0, h1 };
    for (int t = 0; t < MAXLEN; ++t) {
        aec_stepA<<<BATCH, 256, 0, stream>>>(tokens, hbuf[t & 1], low, upp, out, t);
        aec_stepB<<<128, 256, 0, stream>>>(hbuf[t & 1], c, Whp, Wxp, bias, emb,
                                           tokens, hbuf[(t + 1) & 1], t);
    }
}
